// SentenceVAE_4363686773372
// MI455X (gfx1250) — compile-verified, matrix operand order checked by reading, not run on hardware
//
#include <hip/hip_runtime.h>
#include <hip/hip_bf16.h>

// ---------------- problem constants (match reference) ----------------
#define VOCAB  32000
#define DIMK   512
#define ZDIM   64
#define HID    512
#define BATCH  32
#define TLEN   128
#define ROWS   (BATCH*TLEN)     // 4096
#define GATES  (4*HID)          // 2048
#define NSPLIT 125              // VOCAB / 256

typedef __attribute__((ext_vector_type(16))) __bf16 v16bf;
typedef __attribute__((ext_vector_type(8)))  float  v8f;
using u16 = unsigned short;

// ---------------- small helpers ----------------
__device__ __forceinline__ u16 f2bf(float f) {
    unsigned u = __float_as_uint(f);
    unsigned r = u + 0x7FFFu + ((u >> 16) & 1u);   // round-to-nearest-even
    return (u16)(r >> 16);
}
__device__ __forceinline__ float bf2f(u16 h) { return __uint_as_float(((unsigned)h) << 16); }
__device__ __forceinline__ float sigm(float x) { return 1.f / (1.f + __expf(-x)); }

__device__ __forceinline__ v8f wmma_bf16(v16bf a, v16bf b, v8f c) {
    // D(f32,16x16) = A(bf16,16x32) * B(bf16,32x16) + C
    return __builtin_amdgcn_wmma_f32_16x16x32_bf16(false, a, false, b, (short)0, c, false, false);
}

// CDNA5 async memory->LDS path (ASYNCcnt): copies 16B directly into LDS,
// bypassing VGPRs. lds_ptr's low 32 bits are the in-allocation LDS byte offset
// (flat-address aperture rule), which is what the VDST operand wants.
__device__ __forceinline__ void async_ld16(void* lds_ptr, const void* gptr) {
    unsigned ldsoff = (unsigned)(size_t)lds_ptr;
    asm volatile("global_load_async_to_lds_b128 %0, %1, off"
                 :: "v"(ldsoff), "v"(gptr) : "memory");
}
__device__ __forceinline__ void wait_async0() {
    asm volatile("s_wait_asynccnt 0x0" ::: "memory");
}

// Load a 16x32 bf16 A/B fragment from a row-major matrix.
// ISA layout: lane holds row/col = lane%16; K chunks: kb=(lane/16)*8 -> K[kb..kb+7], K[kb+16..kb+23]
__device__ __forceinline__ v16bf ld_frag(const u16* base, int ld) {
    int lane = threadIdx.x & 31;
    int r  = lane & 15;
    int kb = (lane >> 4) << 3;
    const u16* p = base + (size_t)r * ld + kb;
    union { v16bf v; uint4 q[2]; } t;
    t.q[0] = *(const uint4*)(p);
    t.q[1] = *(const uint4*)(p + 16);
    return t.v;
}

// C/D fragment <-> row-major f32:  element (r-th VGPR, lane) = (row r + 8*(lane/16), col lane%16)
__device__ __forceinline__ v8f ld_C(const float* base, int ld) {
    int lane = threadIdx.x & 31;
    int col = lane & 15, hl = lane >> 4;
    v8f c;
#pragma unroll
    for (int r = 0; r < 8; ++r) c[r] = base[(size_t)(r + 8 * hl) * ld + col];
    return c;
}
__device__ __forceinline__ void st_C(float* base, int ld, v8f c) {
    int lane = threadIdx.x & 31;
    int col = lane & 15, hl = lane >> 4;
#pragma unroll
    for (int r = 0; r < 8; ++r) base[(size_t)(r + 8 * hl) * ld + col] = c[r];
}

// ---------------- utility kernels ----------------
__global__ void k_init(unsigned* bars, float* accs, u16* hbuf_f, u16* hbuf_b) {
    int i = blockIdx.x * 256 + threadIdx.x;
    if (i < 4) bars[i] = 0u;
    if (i < 2) accs[i] = 0.f;
    if (i < 2 * BATCH * HID) { hbuf_f[i] = 0; hbuf_b[i] = 0; }
}

__global__ void k_f32_to_bf16(const float* __restrict__ src, u16* __restrict__ dst, int n) {
    int i = blockIdx.x * 256 + threadIdx.x;
    if (i < n) dst[i] = f2bf(src[i]);
}

// Gather embeddings into [T, B, D] bf16 (row = t*B + b), optionally time-flipped.
__global__ void k_gather(const float* __restrict__ emb, const int* __restrict__ x,
                         u16* __restrict__ E, int flip) {
    long long i = (long long)blockIdx.x * 256 + threadIdx.x;
    if (i >= (long long)ROWS * DIMK) return;
    int row = (int)(i >> 9), k = (int)(i & (DIMK - 1));
    int t = row >> 5, b = row & 31;
    int tt = flip ? (TLEN - 1 - t) : t;
    int tok = x[b * TLEN + tt];
    E[i] = f2bf(emb[(size_t)tok * DIMK + k]);
}

// ---------------- input-gate pre-GEMM: G[T*B, 4H] = E @ Wih^T + (bih + bhh) ----------------
// grid(8, 64): x = N-block of 256 gate cols, y = M-block of 64 rows. block = 256 thr (8 waves).
__global__ void k_gates_gemm(const u16* __restrict__ E,   // [ROWS, DIMK] bf16
                             const u16* __restrict__ Wih, // [GATES, DIMK] bf16
                             const float* __restrict__ bih, const float* __restrict__ bhh,
                             float* __restrict__ G) {     // [ROWS, GATES] f32
    extern __shared__ char smem_g[];
    u16* sh_a = (u16*)smem_g;                 // 64 x 512 bf16 = 64 KB
    const int tid = threadIdx.x;
    const int cb = blockIdx.x * 256;
    const int mbase = blockIdx.y * 64;

    // stage A block into LDS via async (memory -> LDS, no VGPR round-trip)
    const char* srcb = (const char*)(E + (size_t)mbase * DIMK);
    for (int i = tid; i < 64 * DIMK / 8; i += 256)
        async_ld16((char*)sh_a + (size_t)i * 16, srcb + (size_t)i * 16);

    const int w = tid >> 5, ln = tid & 15;
    const int n0 = cb + (w * 2) * 16, n1 = n0 + 16;

    v8f acc[4][2];
    float bv0 = bih[n0 + ln] + bhh[n0 + ln];
    float bv1 = bih[n1 + ln] + bhh[n1 + ln];
#pragma unroll
    for (int mt = 0; mt < 4; ++mt)
#pragma unroll
        for (int r = 0; r < 8; ++r) { acc[mt][0][r] = bv0; acc[mt][1][r] = bv1; }

    wait_async0();
    __syncthreads();

    for (int kk = 0; kk < DIMK; kk += 32) {
        v16bf b0 = ld_frag(Wih + (size_t)n0 * DIMK + kk, DIMK);
        v16bf b1 = ld_frag(Wih + (size_t)n1 * DIMK + kk, DIMK);
#pragma unroll
        for (int mt = 0; mt < 4; ++mt) {
            v16bf a = ld_frag(sh_a + (size_t)(mt * 16) * DIMK + kk, DIMK);
            acc[mt][0] = wmma_bf16(a, b0, acc[mt][0]);
            acc[mt][1] = wmma_bf16(a, b1, acc[mt][1]);
        }
    }
#pragma unroll
    for (int mt = 0; mt < 4; ++mt) {
        st_C(G + (size_t)(mbase + mt * 16) * GATES + n0, GATES, acc[mt][0]);
        st_C(G + (size_t)(mbase + mt * 16) * GATES + n1, GATES, acc[mt][1]);
    }
}

// ---------------- persistent recurrent LSTM kernel ----------------
// grid = 16 workgroups (one WGP each), each owns 32 h-columns. Per step:
//   gates[:, owned] = G[t] + h_prev @ Whh^T  (WMMA, K=512)  -> LSTM cell update.
// h ping-pongs through L2 as bf16; per-step sync = atomic grid barrier + cluster barrier.
// The 32KB h stage uses the async->LDS path and overlaps with the G[t] fragment loads.
__global__ void k_lstm_rec(const float* __restrict__ G,    // [ROWS, GATES]
                           const u16* __restrict__ Whh,    // [GATES, HID] bf16
                           u16* __restrict__ hbuf,         // [2, BATCH, HID] bf16 ping-pong
                           float* __restrict__ hfinal,     // [BATCH, HID] f32 (last h)
                           u16* __restrict__ seq_out,      // [ROWS, HID] bf16 or nullptr
                           unsigned* __restrict__ bar) {
    __shared__ u16  sh_h[BATCH * HID];       // 32 KB  (full h_prev, bf16)
    __shared__ float sh_g[4 * BATCH * 32];   // 16 KB  (i,f,g,o for owned 32 cols)
    __shared__ float sh_c[BATCH * 32];       // 4 KB   (owned cell state, persistent)

    const int tid = threadIdx.x;
    const int colbase = blockIdx.x * 32;
    const int w = tid >> 5;
    const int g = w >> 1, nt = w & 1;
    const int gatecol = g * HID + colbase + nt * 16;

    for (int i = tid; i < BATCH * 32; i += 256) sh_c[i] = 0.f;

    for (int t = 0; t < TLEN; ++t) {
        const u16* hrd = hbuf + (size_t)(t & 1) * (BATCH * HID);
        u16* hwr = hbuf + (size_t)((t & 1) ^ 1) * (BATCH * HID);

        // async-stage h_prev into LDS (overlaps with accumulator-fragment loads below)
        for (int i = tid; i < BATCH * HID / 8; i += 256)
            async_ld16((char*)sh_h + (size_t)i * 16, (const char*)hrd + (size_t)i * 16);

        const float* Gt = G + (size_t)(t * BATCH) * GATES;
        v8f acc0 = ld_C(Gt + gatecol, GATES);                       // rows b=0..15
        v8f acc1 = ld_C(Gt + (size_t)16 * GATES + gatecol, GATES);  // rows b=16..31

        wait_async0();
        __syncthreads();

        for (int kk = 0; kk < HID; kk += 32) {
            v16bf bfrag = ld_frag(Whh + (size_t)gatecol * HID + kk, HID);
            v16bf a0 = ld_frag(sh_h + kk, HID);
            v16bf a1 = ld_frag(sh_h + (size_t)16 * HID + kk, HID);
            acc0 = wmma_bf16(a0, bfrag, acc0);
            acc1 = wmma_bf16(a1, bfrag, acc1);
        }
        st_C(sh_g + g * (BATCH * 32) + nt * 16, 32, acc0);
        st_C(sh_g + g * (BATCH * 32) + 16 * 32 + nt * 16, 32, acc1);
        __syncthreads();

        for (int idx = tid; idx < BATCH * 32; idx += 256) {
            int bb = idx >> 5, cc = idx & 31;
            float vi = sh_g[0 * (BATCH * 32) + bb * 32 + cc];
            float vf = sh_g[1 * (BATCH * 32) + bb * 32 + cc];
            float vg = sh_g[2 * (BATCH * 32) + bb * 32 + cc];
            float vo = sh_g[3 * (BATCH * 32) + bb * 32 + cc];
            float c = sigm(vf) * sh_c[idx] + sigm(vi) * tanhf(vg);
            float h = sigm(vo) * tanhf(c);
            sh_c[idx] = c;
            hwr[bb * HID + colbase + cc] = f2bf(h);
            hfinal[bb * HID + colbase + cc] = h;
            if (seq_out) seq_out[(size_t)(t * BATCH + bb) * HID + colbase + cc] = f2bf(h);
        }

        // per-step sync across the 16 workgroups
        __threadfence();
        __syncthreads();
        if (tid == 0) {
            atomicAdd(bar, 1u);
            unsigned target = 16u * (unsigned)(t + 1);
            while (atomicAdd(bar, 0u) < target) __builtin_amdgcn_s_sleep(1);
        }
        __builtin_amdgcn_s_cluster_barrier();   // CDNA5 cluster barrier (NOP outside a cluster)
        __syncthreads();
        __threadfence();
    }
}

// ---------------- latent path (tiny) ----------------
__global__ void k_enc_linh(const float* __restrict__ hf, const float* __restrict__ hb,
                           const float* __restrict__ W, const float* __restrict__ b,
                           float* __restrict__ out) {
    int i = blockIdx.x * 256 + threadIdx.x;
    if (i >= BATCH * HID) return;
    int bb = i >> 9, j = i & (HID - 1);
    const float* wr = W + (size_t)j * (2 * HID);
    const float* a0 = hf + bb * HID;
    const float* a1 = hb + bb * HID;
    float s = b[j];
    for (int k = 0; k < HID; ++k) s += a0[k] * wr[k];
    for (int k = 0; k < HID; ++k) s += a1[k] * wr[HID + k];
    out[i] = s;
}

__global__ void k_latent(const float* __restrict__ henc,
                         const float* __restrict__ meanW, const float* __restrict__ meanb,
                         const float* __restrict__ stdW,  const float* __restrict__ stdb,
                         const float* __restrict__ eps, float* __restrict__ z,
                         float* __restrict__ kl_acc) {
    int i = blockIdx.x * 256 + threadIdx.x;
    if (i >= BATCH * ZDIM) return;
    int bb = i >> 6, j = i & (ZDIM - 1);
    const float* a = henc + bb * HID;
    const float* wm = meanW + (size_t)j * HID;
    const float* ws2 = stdW + (size_t)j * HID;
    float mu = meanb[j], sp = stdb[j];
    for (int k = 0; k < HID; ++k) { mu += a[k] * wm[k]; sp += a[k] * ws2[k]; }
    float sigma = (sp > 20.f) ? sp : log1pf(__expf(sp));     // softplus
    z[i] = mu + sigma * eps[j];
    float kle = -0.5f * (1.f + logf(sigma) - mu * mu - sigma);
    atomicAdd(kl_acc, kle);
}

__global__ void k_dec_init(const float* __restrict__ z, const float* __restrict__ W,
                           const float* __restrict__ b, u16* __restrict__ hbuf0) {
    int i = blockIdx.x * 256 + threadIdx.x;
    if (i >= BATCH * HID) return;
    int bb = i >> 9, j = i & (HID - 1);
    const float* wr = W + (size_t)j * ZDIM;
    const float* a = z + bb * ZDIM;
    float s = b[j];
    for (int k = 0; k < ZDIM; ++k) s += a[k] * wr[k];
    hbuf0[i] = f2bf(tanhf(s));   // dh = tanh(z @ W^T + b), decoder c0 = 0
}

// ---------------- fused logits GEMM + online log-softmax partials ----------------
// grid(NSPLIT, 64): x = 256-vocab-column split, y = 64-row block. block = 256 thr.
// Never materializes logits: emits per-row (max, sumexp) partials for each split.
__global__ void k_logits_softmax(const u16* __restrict__ Hd, // [ROWS, HID] bf16 (dec h)
                                 const u16* __restrict__ Wo, // [VOCAB, HID] bf16
                                 const float* __restrict__ bo,
                                 float2* __restrict__ part) { // [ROWS][NSPLIT]
    extern __shared__ char smem_l[];
    u16* sh_a   = (u16*)smem_l;                        // 64 x 512 bf16 = 64 KB
    float* sh_m = (float*)(smem_l + 64 * HID * 2);     // 8 x 64
    float* sh_s = sh_m + 8 * 64;                       // 8 x 64
    const int tid = threadIdx.x;
    const int cb = blockIdx.x * 256;
    const int mbase = blockIdx.y * 64;

    // async-stage A block (dec_h rows) into LDS
    const char* srcb = (const char*)(Hd + (size_t)mbase * HID);
    for (int i = tid; i < 64 * HID / 8; i += 256)
        async_ld16((char*)sh_a + (size_t)i * 16, srcb + (size_t)i * 16);

    const int w = tid >> 5, ln = tid & 15, hl = (tid & 31) >> 4;
    const int n0 = cb + (w * 2) * 16, n1 = n0 + 16;

    v8f acc[4][2];
    float b0v = bo[n0 + ln], b1v = bo[n1 + ln];
#pragma unroll
    for (int mt = 0; mt < 4; ++mt)
#pragma unroll
        for (int r = 0; r < 8; ++r) { acc[mt][0][r] = b0v; acc[mt][1][r] = b1v; }

    wait_async0();
    __syncthreads();

    for (int kk = 0; kk < HID; kk += 32) {
        __builtin_prefetch(Wo + (size_t)n0 * HID + kk + 128, 0, 1);
        v16bf bf0 = ld_frag(Wo + (size_t)n0 * HID + kk, HID);
        v16bf bf1 = ld_frag(Wo + (size_t)n1 * HID + kk, HID);
#pragma unroll
        for (int mt = 0; mt < 4; ++mt) {
            v16bf a = ld_frag(sh_a + (size_t)(mt * 16) * HID + kk, HID);
            acc[mt][0] = wmma_bf16(a, bf0, acc[mt][0]);
            acc[mt][1] = wmma_bf16(a, bf1, acc[mt][1]);
        }
    }

    // online (max, sumexp) over this block's 256 columns, per row
#pragma unroll
    for (int mt = 0; mt < 4; ++mt) {
#pragma unroll
        for (int r = 0; r < 8; ++r) {
            float v0 = acc[mt][0][r], v1 = acc[mt][1][r];
            float m = fmaxf(v0, v1);
            float s = __expf(v0 - m) + __expf(v1 - m);
#pragma unroll
            for (int d = 1; d < 16; d <<= 1) {   // reduce across 16 lanes of same row-half
                float mo = __shfl_xor(m, d, 32);
                float so = __shfl_xor(s, d, 32);
                float mn = fmaxf(m, mo);
                s = s * __expf(m - mn) + so * __expf(mo - mn);
                m = mn;
            }
            if (ln == 0) {
                int row = mt * 16 + r + 8 * hl;
                sh_m[w * 64 + row] = m;
                sh_s[w * 64 + row] = s;
            }
        }
    }
    __syncthreads();
    if (tid < 64) {
        float M = -__builtin_inff(), S = 0.f;
        for (int w2 = 0; w2 < 8; ++w2) {
            float m = sh_m[w2 * 64 + tid], s = sh_s[w2 * 64 + tid];
            float mn = fmaxf(M, m);
            S = S * __expf(M - mn) + s * __expf(m - mn);
            M = mn;
        }
        part[(size_t)(mbase + tid) * NSPLIT + blockIdx.x] = make_float2(M, S);
    }
}

// logit at the target token per row (tiny: 4096 x 512 dot)
__global__ void k_target_logit(const u16* __restrict__ Hd, const float* __restrict__ Wo_f,
                               const float* __restrict__ bo, const int* __restrict__ x,
                               float* __restrict__ tgt) {
    int row = blockIdx.x * 256 + threadIdx.x;
    if (row >= ROWS) return;
    int b = row & 31, t = row >> 5;
    int tok = x[b * TLEN + t];
    const u16* h = Hd + (size_t)row * HID;
    const float* wr = Wo_f + (size_t)tok * HID;
    float s = bo[tok];
    for (int k = 0; k < HID; ++k) s += bf2f(h[k]) * wr[k];
    tgt[row] = s;
}

__global__ void k_reduce_rows(const float2* __restrict__ part, const float* __restrict__ tgt,
                              float* __restrict__ recon_acc) {
    int row = blockIdx.x * 256 + threadIdx.x;
    if (row >= ROWS) return;
    float M = -__builtin_inff(), S = 0.f;
    for (int i = 0; i < NSPLIT; ++i) {
        float2 p = part[(size_t)row * NSPLIT + i];
        float mn = fmaxf(M, p.x);
        S = S * __expf(M - mn) + p.y * __expf(p.x - mn);
        M = mn;
    }
    float lse = M + logf(S);
    atomicAdd(recon_acc, lse - tgt[row]);   // -logp[target]
}

__global__ void k_finalize(const float* __restrict__ accs, float* __restrict__ out) {
    if (threadIdx.x == 0 && blockIdx.x == 0) out[0] = (accs[0] + accs[1]) / (float)BATCH;
}

// ---------------- host orchestration ----------------
extern "C" void kernel_launch(void* const* d_in, const int* in_sizes, int n_in,
                              void* d_out, int out_size, void* d_ws, size_t ws_size,
                              hipStream_t stream) {
    (void)in_sizes; (void)n_in; (void)out_size; (void)ws_size;
    const int*   x        = (const int*)  d_in[0];
    const float* eps      = (const float*)d_in[1];
    const float* enc_emb  = (const float*)d_in[2];
    const float* dec_emb  = (const float*)d_in[3];
    const float* f_Wih = (const float*)d_in[4];  const float* f_Whh = (const float*)d_in[5];
    const float* f_bih = (const float*)d_in[6];  const float* f_bhh = (const float*)d_in[7];
    const float* b_Wih = (const float*)d_in[8];  const float* b_Whh = (const float*)d_in[9];
    const float* b_bih = (const float*)d_in[10]; const float* b_bhh = (const float*)d_in[11];
    const float* d_Wih = (const float*)d_in[12]; const float* d_Whh = (const float*)d_in[13];
    const float* d_bih = (const float*)d_in[14]; const float* d_bhh = (const float*)d_in[15];
    const float* enc_linh_W = (const float*)d_in[16]; const float* enc_linh_b = (const float*)d_in[17];
    const float* mean_W = (const float*)d_in[18]; const float* mean_b = (const float*)d_in[19];
    const float* std_W  = (const float*)d_in[20]; const float* std_b  = (const float*)d_in[21];
    const float* dec_linh_W = (const float*)d_in[22]; const float* dec_linh_b = (const float*)d_in[23];
    const float* out_W = (const float*)d_in[24]; const float* out_b = (const float*)d_in[25];
    float* out = (float*)d_out;

    // ---- workspace carve ----
    char* p = (char*)d_ws;
    auto alloc = [&](size_t bytes) -> char* {
        char* r = p; p += (bytes + 255) & ~(size_t)255; return r;
    };
    const size_t WSZ = (size_t)GATES * DIMK * sizeof(u16);   // Wih/Whh bf16: 2 MB each
    u16* fWih_h = (u16*)alloc(WSZ); u16* fWhh_h = (u16*)alloc(WSZ);
    u16* bWih_h = (u16*)alloc(WSZ); u16* bWhh_h = (u16*)alloc(WSZ);
    u16* dWih_h = (u16*)alloc(WSZ); u16* dWhh_h = (u16*)alloc(WSZ);
    u16* Wo_h   = (u16*)alloc((size_t)VOCAB * HID * sizeof(u16));       // 31 MB (L2-resident)
    u16* Ef = (u16*)alloc((size_t)ROWS * DIMK * sizeof(u16));
    u16* Eb = (u16*)alloc((size_t)ROWS * DIMK * sizeof(u16));
    u16* Ed = (u16*)alloc((size_t)ROWS * DIMK * sizeof(u16));
    float* Gf = (float*)alloc((size_t)ROWS * GATES * sizeof(float));    // 32 MB each
    float* Gb = (float*)alloc((size_t)ROWS * GATES * sizeof(float));
    float* Gd = (float*)alloc((size_t)ROWS * GATES * sizeof(float));
    u16* dec_h = (u16*)alloc((size_t)ROWS * HID * sizeof(u16));
    u16* hbuf_f = (u16*)alloc((size_t)2 * BATCH * HID * sizeof(u16));
    u16* hbuf_b = (u16*)alloc((size_t)2 * BATCH * HID * sizeof(u16));
    u16* hbuf_d = (u16*)alloc((size_t)2 * BATCH * HID * sizeof(u16));
    float* hfin_f = (float*)alloc((size_t)BATCH * HID * sizeof(float));
    float* hfin_b = (float*)alloc((size_t)BATCH * HID * sizeof(float));
    float* hfin_scratch = (float*)alloc((size_t)BATCH * HID * sizeof(float));
    float* henc = (float*)alloc((size_t)BATCH * HID * sizeof(float));
    float* zlat = (float*)alloc((size_t)BATCH * ZDIM * sizeof(float));
    float2* part = (float2*)alloc((size_t)ROWS * NSPLIT * sizeof(float2)); // 3.9 MB
    float* tgt = (float*)alloc((size_t)ROWS * sizeof(float));
    unsigned* bars = (unsigned*)alloc(4 * sizeof(unsigned));
    float* accs = (float*)alloc(2 * sizeof(float));   // [0]=kl, [1]=recon

    // ---- pipeline ----
    k_init<<<128, 256, 0, stream>>>(bars, accs, hbuf_f, hbuf_b);

    auto cvt = [&](const float* s, u16* d, int n) {
        k_f32_to_bf16<<<(n + 255) / 256, 256, 0, stream>>>(s, d, n);
    };
    cvt(f_Wih, fWih_h, GATES * DIMK); cvt(f_Whh, fWhh_h, GATES * HID);
    cvt(b_Wih, bWih_h, GATES * DIMK); cvt(b_Whh, bWhh_h, GATES * HID);
    cvt(d_Wih, dWih_h, GATES * DIMK); cvt(d_Whh, dWhh_h, GATES * HID);
    cvt(out_W, Wo_h, VOCAB * HID);

    const int ngat = (ROWS * DIMK + 255) / 256;
    k_gather<<<ngat, 256, 0, stream>>>(enc_emb, x, Ef, 0);
    k_gather<<<ngat, 256, 0, stream>>>(enc_emb, x, Eb, 1);
    k_gather<<<ngat, 256, 0, stream>>>(dec_emb, x, Ed, 0);

    dim3 ggrid(GATES / 256, ROWS / 64);
    size_t glds = (size_t)64 * DIMK * sizeof(u16);
    k_gates_gemm<<<ggrid, 256, glds, stream>>>(Ef, fWih_h, f_bih, f_bhh, Gf);
    k_gates_gemm<<<ggrid, 256, glds, stream>>>(Eb, bWih_h, b_bih, b_bhh, Gb);
    k_gates_gemm<<<ggrid, 256, glds, stream>>>(Ed, dWih_h, d_bih, d_bhh, Gd);

    // encoder LSTMs (persistent, 16 WGPs each, per-step grid+cluster barrier)
    k_lstm_rec<<<16, 256, 0, stream>>>(Gf, fWhh_h, hbuf_f, hfin_f, nullptr, bars + 0);
    k_lstm_rec<<<16, 256, 0, stream>>>(Gb, bWhh_h, hbuf_b, hfin_b, nullptr, bars + 1);

    // latent path
    k_enc_linh<<<(BATCH * HID + 255) / 256, 256, 0, stream>>>(hfin_f, hfin_b, enc_linh_W, enc_linh_b, henc);
    k_latent<<<(BATCH * ZDIM + 255) / 256, 256, 0, stream>>>(henc, mean_W, mean_b, std_W, std_b, eps, zlat, accs + 0);
    k_dec_init<<<(BATCH * HID + 255) / 256, 256, 0, stream>>>(zlat, dec_linh_W, dec_linh_b, hbuf_d);

    // decoder LSTM (stores all h_t as bf16 for the logits GEMM)
    k_lstm_rec<<<16, 256, 0, stream>>>(Gd, dWhh_h, hbuf_d, hfin_scratch, dec_h, bars + 2);

    // fused logits GEMM + log-softmax + NLL
    k_target_logit<<<(ROWS + 255) / 256, 256, 0, stream>>>(dec_h, out_W, out_b, x, tgt);
    dim3 lgrid(NSPLIT, ROWS / 64);
    size_t llds = (size_t)64 * HID * sizeof(u16) + 2 * 8 * 64 * sizeof(float);
    k_logits_softmax<<<lgrid, 256, llds, stream>>>(dec_h, Wo_h, out_b, part);
    k_reduce_rows<<<(ROWS + 255) / 256, 256, 0, stream>>>(part, tgt, accs + 1);
    k_finalize<<<1, 64, 0, stream>>>(accs, out);
}
